// EGNNLayer_33672543601320
// MI455X (gfx1250) — matrix operand order repaired and checked by reading, not compiled, not run on hardware
//
#include <hip/hip_runtime.h>
#include <hip/hip_bf16.h>

#define FEAT 128
#define LN_EPS 1e-5f

typedef __attribute__((ext_vector_type(16))) __bf16 v16bf;
typedef __attribute__((ext_vector_type(8)))  float  v8f;

union Frag16 {
    v16bf v;
    __bf16 h[16];
    unsigned int u[8];
};

union PkBf {
    __bf16 h[2];
    unsigned int u;
};

__device__ __forceinline__ v8f wmma_bf16(const Frag16& a, const Frag16& b, v8f c) {
    return __builtin_amdgcn_wmma_f32_16x16x32_bf16(false, a.v, false, b.v,
                                                   (short)0, c, false, false);
}

// Pack a (kblocks*32) x 128 f32 weight into per-fragment bf16 B layout:
// packed[kb][nb][lane][d]; lanes 0-15 hold K {0..7,16..23}, lanes 16-31 hold K {8..15,24..31}
// of the 32-K block; dword d packs K = klo+2d(+8 if d>=4), +1 at column nb*16+lane%16.
__device__ __forceinline__ void pack_weight_b(const float* __restrict__ w,
                                              unsigned int* __restrict__ packed,
                                              int kblocks, int tid, int nthreads) {
    const int total = kblocks * 8 * 32 * 8;
    for (int i = tid; i < total; i += nthreads) {
        int d    = i & 7;
        int lane = (i >> 3) & 31;
        int nb   = (i >> 8) & 7;
        int kb   = i >> 11;
        int K  = kb * 32 + ((lane < 16) ? 0 : 8) + 2 * d + ((d >= 4) ? 8 : 0);
        int Nc = nb * 16 + (lane & 15);
        PkBf p;
        p.h[0] = (__bf16)w[(size_t)K * FEAT + Nc];
        p.h[1] = (__bf16)w[(size_t)(K + 1) * FEAT + Nc];
        packed[i] = p.u;
    }
}

__device__ __forceinline__ void load_b_frag(Frag16& b, const unsigned int* __restrict__ packed,
                                            int kb, int nb, int lane) {
    const uint4* q = (const uint4*)(packed + (((kb * 8 + nb) * 32 + lane) << 3));
    uint4 q0 = q[0];
    uint4 q1 = q[1];
    b.u[0] = q0.x; b.u[1] = q0.y; b.u[2] = q0.z; b.u[3] = q0.w;
    b.u[4] = q1.x; b.u[5] = q1.y; b.u[6] = q1.z; b.u[7] = q1.w;
}

// A fragment (16x32 bf16) from an f32 row in global memory.
// Lanes 0-15: K = kb*32 + {0..7, 16..23}; lanes 16-31: K = kb*32 + {8..15, 24..31}.
__device__ __forceinline__ void load_a_f32(Frag16& a, const float* __restrict__ rowp,
                                           int kb, int lane) {
    int klo = kb * 32 + ((lane & 16) ? 8 : 0);
    const float4* p0 = (const float4*)(rowp + klo);
    const float4* p1 = (const float4*)(rowp + klo + 16);
    float4 f0 = p0[0], f1 = p0[1], f2 = p1[0], f3 = p1[1];
    a.h[0]  = (__bf16)f0.x;  a.h[1]  = (__bf16)f0.y;
    a.h[2]  = (__bf16)f0.z;  a.h[3]  = (__bf16)f0.w;
    a.h[4]  = (__bf16)f1.x;  a.h[5]  = (__bf16)f1.y;
    a.h[6]  = (__bf16)f1.z;  a.h[7]  = (__bf16)f1.w;
    a.h[8]  = (__bf16)f2.x;  a.h[9]  = (__bf16)f2.y;
    a.h[10] = (__bf16)f2.z;  a.h[11] = (__bf16)f2.w;
    a.h[12] = (__bf16)f3.x;  a.h[13] = (__bf16)f3.y;
    a.h[14] = (__bf16)f3.z;  a.h[15] = (__bf16)f3.w;
}

// A fragment from a bf16 row in LDS.
__device__ __forceinline__ void load_a_lds(Frag16& a, const __bf16* __restrict__ rowp,
                                           int kb, int lane) {
    int klo = kb * 32 + ((lane & 16) ? 8 : 0);
    uint4 q0 = *(const uint4*)(rowp + klo);
    uint4 q1 = *(const uint4*)(rowp + klo + 16);
    a.u[0] = q0.x; a.u[1] = q0.y; a.u[2] = q0.z; a.u[3] = q0.w;
    a.u[4] = q1.x; a.u[5] = q1.y; a.u[6] = q1.z; a.u[7] = q1.w;
}

__device__ __forceinline__ float silu(float v) {
    return v / (1.0f + __expf(-v));
}

// ---------------------------------------------------------------------------
// init: agg = 0, pos_out = pos
__global__ void egnn_init_kernel(const float* __restrict__ pos,
                                 float* __restrict__ agg,
                                 float* __restrict__ posout,
                                 int nAgg, int nPos) {
    int i = blockIdx.x * blockDim.x + threadIdx.x;
    int stride = gridDim.x * blockDim.x;
    for (int j = i; j < nAgg; j += stride) agg[j] = 0.0f;
    for (int j = i; j < nPos; j += stride) posout[j] = pos[j];
}

// ---------------------------------------------------------------------------
// Edge kernel: msg = silu([x_r, x_c, d2] @ ew1 + eb1) @ ew2 + eb2
// atomically scatters msg into agg[row], tanh(msg@cw+cb)*diff into posout[row].
#define EDGE_SMEM ((16384 + 8192) * 4 + 8 * 16 * FEAT * 2)   // 131072 B

__global__ void __launch_bounds__(256)
egnn_edge_kernel(const float* __restrict__ x,
                 const float* __restrict__ pos,
                 const int*   __restrict__ ei,
                 const float* __restrict__ ew1,
                 const float* __restrict__ eb1,
                 const float* __restrict__ ew2,
                 const float* __restrict__ eb2,
                 const float* __restrict__ cw,
                 const float* __restrict__ cb,
                 float* __restrict__ agg,
                 float* __restrict__ posout,
                 int Ne) {
    extern __shared__ char smem[];
    unsigned int* ew1p = (unsigned int*)smem;            // 8 kb * 8 nb * 32 * 8 dwords
    unsigned int* ew2p = ew1p + 16384;                   // 4 kb * 8 nb * 32 * 8
    __bf16* hbase = (__bf16*)(ew2p + 8192);

    pack_weight_b(ew1, ew1p, 8, threadIdx.x, blockDim.x);   // K rows 0..255 of ew1
    pack_weight_b(ew2, ew2p, 4, threadIdx.x, blockDim.x);
    __syncthreads();

    const int lane = threadIdx.x & 31;
    const int wave = threadIdx.x >> 5;
    const int gwave = blockIdx.x * (blockDim.x >> 5) + wave;
    const int nwaves = gridDim.x * (blockDim.x >> 5);
    const int ntiles = (Ne + 15) >> 4;
    __bf16* myscr = hbase + wave * (16 * FEAT);

    const int m_l  = lane & 15;
    const int mrow = (lane & 16) ? 8 : 0;   // C-fragment: VGPR r holds row r+mrow

    for (int tile = gwave; tile < ntiles; tile += nwaves) {
        int e = tile * 16 + m_l;
        int ec = (e < Ne) ? e : 0;
        int rowi = ei[ec];
        int coli = ei[Ne + ec];
        float dx = pos[(size_t)rowi * 3 + 0] - pos[(size_t)coli * 3 + 0];
        float dy = pos[(size_t)rowi * 3 + 1] - pos[(size_t)coli * 3 + 1];
        float dz = pos[(size_t)rowi * 3 + 2] - pos[(size_t)coli * 3 + 2];
        float d2 = dx * dx + dy * dy + dz * dz;
        const float* xr = x + (size_t)rowi * FEAT;
        const float* xc = x + (size_t)coli * FEAT;

        // Per-accumulator-row broadcasts (rows this lane's C fragment covers).
        float d2b[8];
        int   rmB[8];
        int   vldB[8];
#pragma unroll
        for (int r = 0; r < 8; ++r) {
            int mm = r + mrow;
            d2b[r]  = __shfl(d2, mm, 32);
            rmB[r]  = __shfl(rowi, mm, 32);
            vldB[r] = (tile * 16 + mm) < Ne;
        }

        // A fragments are N-invariant: build once per tile, reuse for all 8 nb.
        Frag16 aR[4], aC[4];
#pragma unroll
        for (int kb = 0; kb < 4; ++kb) {
            load_a_f32(aR[kb], xr, kb, lane);
            load_a_f32(aC[kb], xc, kb, lane);
        }

        // ---- GEMM1: [x_r | x_c | d2] @ ew1 + eb1, SiLU -> LDS scratch (bf16)
#pragma unroll
        for (int nb = 0; nb < 8; ++nb) {
            int ncol = nb * 16 + m_l;
            v8f acc = {0.f, 0.f, 0.f, 0.f, 0.f, 0.f, 0.f, 0.f};
#pragma unroll
            for (int kb = 0; kb < 4; ++kb) {
                Frag16 b;
                load_b_frag(b, ew1p, kb, nb, lane);
                acc = wmma_bf16(aR[kb], b, acc);
                load_b_frag(b, ew1p, kb + 4, nb, lane);   // rows 128..255 of ew1
                acc = wmma_bf16(aC[kb], b, acc);
            }
            float wlast = ew1[(size_t)256 * FEAT + ncol];  // dist2 rank-1 term (f32)
            float b1 = eb1[ncol];
#pragma unroll
            for (int r = 0; r < 8; ++r) {
                float v = acc[r] + d2b[r] * wlast + b1;
                myscr[(r + mrow) * FEAT + ncol] = (__bf16)silu(v);
            }
        }

        // ---- GEMM2: msg = h @ ew2 + eb2 ; scatter + cw dot
        Frag16 ah[4];
#pragma unroll
        for (int kb = 0; kb < 4; ++kb)
            load_a_lds(ah[kb], myscr + m_l * FEAT, kb, lane);

        float dotr[8] = {0.f, 0.f, 0.f, 0.f, 0.f, 0.f, 0.f, 0.f};
#pragma unroll
        for (int nb = 0; nb < 8; ++nb) {
            int ncol = nb * 16 + m_l;
            v8f acc = {0.f, 0.f, 0.f, 0.f, 0.f, 0.f, 0.f, 0.f};
#pragma unroll
            for (int kb = 0; kb < 4; ++kb) {
                Frag16 b;
                load_b_frag(b, ew2p, kb, nb, lane);
                acc = wmma_bf16(ah[kb], b, acc);
            }
            float b2  = eb2[ncol];
            float cwn = cw[ncol];
#pragma unroll
            for (int r = 0; r < 8; ++r) {
                float msg = acc[r] + b2;
                dotr[r] += msg * cwn;
                if (vldB[r]) atomicAdd(&agg[(size_t)rmB[r] * FEAT + ncol], msg);
            }
        }

        // Reduce msg.cw over the 16 lanes of each half (each half covers rows 0-7 / 8-15).
#pragma unroll
        for (int r = 0; r < 8; ++r) {
            float s = dotr[r];
            s += __shfl_xor(s, 8, 16);
            s += __shfl_xor(s, 4, 16);
            s += __shfl_xor(s, 2, 16);
            s += __shfl_xor(s, 1, 16);
            dotr[r] = s;
        }

        // Coordinate update: lanes 0-7 handle rows 0-7, lanes 16-23 rows 8-15.
        float cbv = cb[0];
        int rsel = lane & 7;
        float dsel = 0.f;
#pragma unroll
        for (int r = 0; r < 8; ++r) dsel = (rsel == r) ? dotr[r] : dsel;
        float sc = tanhf(dsel + cbv);
        int mh = (lane & 7) + mrow;                 // row this lane would handle
        float dxh = __shfl(dx, mh, 32);             // shuffles outside divergence
        float dyh = __shfl(dy, mh, 32);
        float dzh = __shfl(dz, mh, 32);
        int   rmh = __shfl(rowi, mh, 32);
        bool doit = ((lane & 15) < 8) && ((tile * 16 + mh) < Ne);
        if (doit) {
            atomicAdd(&posout[(size_t)rmh * 3 + 0], sc * dxh);
            atomicAdd(&posout[(size_t)rmh * 3 + 1], sc * dyh);
            atomicAdd(&posout[(size_t)rmh * 3 + 2], sc * dzh);
        }
    }
}

// ---------------------------------------------------------------------------
// Node kernel: upd = silu(agg@nw1+nb1)@nw2+nb2 ; x_out = LN(x+upd)*gamma+beta
#define NODE_SMEM ((8192 + 8192) * 4 + 8 * 16 * FEAT * 2)    // 98304 B

__global__ void __launch_bounds__(256)
egnn_node_kernel(const float* __restrict__ x,
                 const float* __restrict__ agg,
                 const float* __restrict__ nw1,
                 const float* __restrict__ nb1,
                 const float* __restrict__ nw2,
                 const float* __restrict__ nb2,
                 const float* __restrict__ gamma,
                 const float* __restrict__ beta,
                 float* __restrict__ xout,
                 int Nn) {
    extern __shared__ char smem[];
    unsigned int* w1p = (unsigned int*)smem;     // 4*8*32*8
    unsigned int* w2p = w1p + 8192;
    __bf16* hbase = (__bf16*)(w2p + 8192);

    pack_weight_b(nw1, w1p, 4, threadIdx.x, blockDim.x);
    pack_weight_b(nw2, w2p, 4, threadIdx.x, blockDim.x);
    __syncthreads();

    const int lane = threadIdx.x & 31;
    const int wave = threadIdx.x >> 5;
    const int gwave = blockIdx.x * (blockDim.x >> 5) + wave;
    const int nwaves = gridDim.x * (blockDim.x >> 5);
    const int ntiles = (Nn + 15) >> 4;
    __bf16* myscr = hbase + wave * (16 * FEAT);

    const int m_l  = lane & 15;
    const int mrow = (lane & 16) ? 8 : 0;

    for (int tile = gwave; tile < ntiles; tile += nwaves) {
        int node = tile * 16 + m_l;
        int nc = (node < Nn) ? node : 0;
        const float* ar = agg + (size_t)nc * FEAT;

        // A fragments for GEMM1 (N-invariant).
        Frag16 aA[4];
#pragma unroll
        for (int kb = 0; kb < 4; ++kb)
            load_a_f32(aA[kb], ar, kb, lane);

        // ---- GEMM1: silu(agg @ nw1 + nb1) -> LDS scratch
#pragma unroll
        for (int nb = 0; nb < 8; ++nb) {
            int ncol = nb * 16 + m_l;
            v8f acc = {0.f, 0.f, 0.f, 0.f, 0.f, 0.f, 0.f, 0.f};
#pragma unroll
            for (int kb = 0; kb < 4; ++kb) {
                Frag16 b;
                load_b_frag(b, w1p, kb, nb, lane);
                acc = wmma_bf16(aA[kb], b, acc);
            }
            float b1 = nb1[ncol];
#pragma unroll
            for (int r = 0; r < 8; ++r)
                myscr[(r + mrow) * FEAT + ncol] = (__bf16)silu(acc[r] + b1);
        }

        // ---- GEMM2 + residual; accumulate LN statistics
        Frag16 ah[4];
#pragma unroll
        for (int kb = 0; kb < 4; ++kb)
            load_a_lds(ah[kb], myscr + m_l * FEAT, kb, lane);

        float pre[8][8];
        float sum[8] = {0.f, 0.f, 0.f, 0.f, 0.f, 0.f, 0.f, 0.f};
        float sq[8]  = {0.f, 0.f, 0.f, 0.f, 0.f, 0.f, 0.f, 0.f};
#pragma unroll
        for (int nb = 0; nb < 8; ++nb) {
            int ncol = nb * 16 + m_l;
            v8f acc = {0.f, 0.f, 0.f, 0.f, 0.f, 0.f, 0.f, 0.f};
#pragma unroll
            for (int kb = 0; kb < 4; ++kb) {
                Frag16 b;
                load_b_frag(b, w2p, kb, nb, lane);
                acc = wmma_bf16(ah[kb], b, acc);
            }
            float b2 = nb2[ncol];
#pragma unroll
            for (int r = 0; r < 8; ++r) {
                int nr = tile * 16 + r + mrow;
                int nrc = (nr < Nn) ? nr : 0;
                float p = x[(size_t)nrc * FEAT + ncol] + acc[r] + b2;
                pre[nb][r] = p;
                sum[r] += p;
                sq[r]  += p * p;
            }
        }

        // ---- LayerNorm stats across 16 lanes per half
        float meanA[8], invA[8];
#pragma unroll
        for (int r = 0; r < 8; ++r) {
            float s = sum[r], q = sq[r];
            s += __shfl_xor(s, 8, 16);  q += __shfl_xor(q, 8, 16);
            s += __shfl_xor(s, 4, 16);  q += __shfl_xor(q, 4, 16);
            s += __shfl_xor(s, 2, 16);  q += __shfl_xor(q, 2, 16);
            s += __shfl_xor(s, 1, 16);  q += __shfl_xor(q, 1, 16);
            float mean = s * (1.0f / FEAT);
            float var  = q * (1.0f / FEAT) - mean * mean;
            meanA[r] = mean;
            invA[r]  = rsqrtf(var + LN_EPS);
        }

        // ---- normalize + affine + store
#pragma unroll
        for (int nb = 0; nb < 8; ++nb) {
            int ncol = nb * 16 + m_l;
            float g  = gamma[ncol];
            float bt = beta[ncol];
#pragma unroll
            for (int r = 0; r < 8; ++r) {
                int nr = tile * 16 + r + mrow;
                if (nr < Nn)
                    xout[(size_t)nr * FEAT + ncol] =
                        g * (pre[nb][r] - meanA[r]) * invA[r] + bt;
            }
        }
    }
}

// ---------------------------------------------------------------------------
extern "C" void kernel_launch(void* const* d_in, const int* in_sizes, int n_in,
                              void* d_out, int out_size, void* d_ws, size_t ws_size,
                              hipStream_t stream) {
    const float* x     = (const float*)d_in[0];
    const float* pos   = (const float*)d_in[1];
    const int*   ei    = (const int*)d_in[2];
    const float* ew1   = (const float*)d_in[3];
    const float* eb1   = (const float*)d_in[4];
    const float* ew2   = (const float*)d_in[5];
    const float* eb2   = (const float*)d_in[6];
    const float* nw1   = (const float*)d_in[7];
    const float* nb1   = (const float*)d_in[8];
    const float* nw2   = (const float*)d_in[9];
    const float* nb2   = (const float*)d_in[10];
    const float* cw    = (const float*)d_in[11];
    const float* cb    = (const float*)d_in[12];
    const float* gamma = (const float*)d_in[13];
    const float* beta  = (const float*)d_in[14];

    int Nn = in_sizes[0] / FEAT;
    int Ne = in_sizes[2] / 2;

    float* xout   = (float*)d_out;
    float* posout = xout + (size_t)Nn * FEAT;
    float* agg    = (float*)d_ws;

    (void)hipFuncSetAttribute((const void*)egnn_edge_kernel,
                              hipFuncAttributeMaxDynamicSharedMemorySize, EDGE_SMEM);
    (void)hipFuncSetAttribute((const void*)egnn_node_kernel,
                              hipFuncAttributeMaxDynamicSharedMemorySize, NODE_SMEM);

    egnn_init_kernel<<<1024, 256, 0, stream>>>(pos, agg, posout, Nn * FEAT, Nn * 3);

    int etiles = (Ne + 15) >> 4;
    int eblocks = (etiles + 7) / 8;
    if (eblocks > 1024) eblocks = 1024;
    egnn_edge_kernel<<<eblocks, 256, EDGE_SMEM, stream>>>(
        x, pos, ei, ew1, eb1, ew2, eb2, cw, cb, agg, posout, Ne);

    int ntls = (Nn + 15) >> 4;
    int nblocks = (ntls + 7) / 8;
    if (nblocks > 512) nblocks = 512;
    egnn_node_kernel<<<nblocks, 256, NODE_SMEM, stream>>>(
        x, agg, nw1, nb1, nw2, nb2, gamma, beta, xout, Nn);
}
